// POSdata_65472481460367
// MI455X (gfx1250) — compile-verified, matrix operand order checked by reading, not run on hardware
//
#include <hip/hip_runtime.h>
#include <cstdint>

#define P_TOTAL   2990
#define NT        32
#define NC        21
#define NTHREADS  1024

// ---- LDS byte-offset layout (dynamic shared, base offset 0) ----
#define OFF_PR     0                          // 2990*4 f32 point-form priors (47840 B)
#define OFF_OV     (P_TOTAL * 16)             // 2990 f32 best_truth_ov
#define OFF_IDX    (OFF_OV + P_TOTAL * 4)     // 2990 i32 best_truth_idx
#define OFF_TRUTH  (OFF_IDX + P_TOTAL * 4)    // 32*4 f32 truths (point form already)
#define OFF_LABEL  (OFF_TRUTH + 512)          // 32 f32 labels
#define OFF_AREA   (OFF_LABEL + 128)          // 32 f32 truth areas
#define OFF_BP     (OFF_AREA + 128)           // 32 i32 best_prior_idx
#define OFF_CNT    (OFF_BP + 128)             // 32 waves * 32 i32 counters
#define SMEM_BYTES (OFF_CNT + 32 * 32 * 4)    // 76752 B  (< 320 KB WGP LDS)

typedef _Float16 v16h __attribute__((ext_vector_type(16)));
typedef float    v8f  __attribute__((ext_vector_type(8)));

__device__ __forceinline__ float iou_box(float ax1, float ay1, float ax2, float ay2,
                                         float areaA,
                                         float bx1, float by1, float bx2, float by2,
                                         float areaB) {
  float iw = fminf(ax2, bx2) - fmaxf(ax1, bx1);
  iw = fmaxf(iw, 0.0f);
  float ih = fminf(ay2, by2) - fmaxf(ay1, by1);
  ih = fmaxf(ih, 0.0f);
  float inter = iw * ih;
  return inter / (areaA + areaB - inter);
}

__global__ __launch_bounds__(NTHREADS) void ssd_match_hist_kernel(
    const float* __restrict__ conf,     // conf_data[0]: [2990,21]
    const float* __restrict__ priors,   // [2990,4] cx,cy,w,h
    const float* __restrict__ targets,  // targets[0]: [32,5]
    float* __restrict__ out)            // [6,4]
{
  extern __shared__ char smem[];
  float* s_pr   = (float*)(smem + OFF_PR);
  float* s_ov   = (float*)(smem + OFF_OV);
  int*   s_idx  = (int*)  (smem + OFF_IDX);
  float* s_tr   = (float*)(smem + OFF_TRUTH);
  float* s_lab  = (float*)(smem + OFF_LABEL);
  float* s_area = (float*)(smem + OFF_AREA);
  int*   s_bp   = (int*)  (smem + OFF_BP);
  int*   s_cnt  = (int*)  (smem + OFF_CNT);

  const int tid = threadIdx.x;

  // ---- Phase 0: async DMA priors -> LDS (ASYNCcnt path), load targets, zero counters
  for (int p = tid; p < P_TOTAL; p += NTHREADS) {
    uint64_t ga = (uint64_t)(uintptr_t)priors + (uint64_t)p * 16u;
    uint32_t la = (uint32_t)(OFF_PR + p * 16);
    asm volatile("global_load_async_to_lds_b128 %0, %1, off"
                 :: "v"(la), "v"(ga) : "memory");
  }
  if (tid < NT * 5) {
    int t = tid / 5, f = tid % 5;
    float v = targets[tid];
    if (f < 4) s_tr[t * 4 + f] = v; else s_lab[t] = v;
  }
  s_cnt[tid] = 0;
  asm volatile("s_wait_asynccnt 0" ::: "memory");
  __syncthreads();

  // truth areas (reference: from truth coords directly)
  if (tid < NT) {
    float x1 = s_tr[tid * 4 + 0], y1 = s_tr[tid * 4 + 1];
    float x2 = s_tr[tid * 4 + 2], y2 = s_tr[tid * 4 + 3];
    s_area[tid] = (x2 - x1) * (y2 - y1);
  }
  // priors -> point form, in place (same thread wrote nothing; async filled it)
  for (int p = tid; p < P_TOTAL; p += NTHREADS) {
    float cx = s_pr[p * 4 + 0], cy = s_pr[p * 4 + 1];
    float w  = s_pr[p * 4 + 2], h  = s_pr[p * 4 + 3];
    s_pr[p * 4 + 0] = cx - w * 0.5f;
    s_pr[p * 4 + 1] = cy - h * 0.5f;
    s_pr[p * 4 + 2] = cx + w * 0.5f;
    s_pr[p * 4 + 3] = cy + h * 0.5f;
  }
  __syncthreads();

  // ---- Phase 1: per prior, max/argmax IoU over 32 truths (first-wins)
  for (int p = tid; p < P_TOTAL; p += NTHREADS) {
    float bx1 = s_pr[p * 4 + 0], by1 = s_pr[p * 4 + 1];
    float bx2 = s_pr[p * 4 + 2], by2 = s_pr[p * 4 + 3];
    float areaB = (bx2 - bx1) * (by2 - by1);   // reference uses point-form diffs
    float best = -1.0f; int bi = 0;
    #pragma unroll 4
    for (int t = 0; t < NT; ++t) {
      float v = iou_box(s_tr[t * 4 + 0], s_tr[t * 4 + 1], s_tr[t * 4 + 2], s_tr[t * 4 + 3],
                        s_area[t], bx1, by1, bx2, by2, areaB);
      if (v > best) { best = v; bi = t; }
    }
    s_ov[p]  = best;
    s_idx[p] = bi;
  }
  __syncthreads();

  // ---- Phase 2: per truth, argmax IoU over priors. Wave w handles truth w.
  {
    const int w = tid >> 5, l = tid & 31;
    float ax1 = s_tr[w * 4 + 0], ay1 = s_tr[w * 4 + 1];
    float ax2 = s_tr[w * 4 + 2], ay2 = s_tr[w * 4 + 3];
    float areaA = s_area[w];
    float best = -1.0f; int bp = P_TOTAL;
    for (int p = l; p < P_TOTAL; p += 32) {
      float bx1 = s_pr[p * 4 + 0], by1 = s_pr[p * 4 + 1];
      float bx2 = s_pr[p * 4 + 2], by2 = s_pr[p * 4 + 3];
      float areaB = (bx2 - bx1) * (by2 - by1);
      float v = iou_box(ax1, ay1, ax2, ay2, areaA, bx1, by1, bx2, by2, areaB);
      if (v > best) { best = v; bp = p; }   // p increases -> first occurrence kept
    }
    // wave32 butterfly: max with smallest-index tie break (jnp.argmax semantics)
    #pragma unroll
    for (int m = 16; m >= 1; m >>= 1) {
      float ob = __shfl_xor(best, m, 32);
      int   op = __shfl_xor(bp,   m, 32);
      if (ob > best || (ob == best && op < bp)) { best = ob; bp = op; }
    }
    if (l == 0) s_bp[w] = bp;
  }
  __syncthreads();

  // ---- Phase 2b: scatter override. ov=2.0; idx: last write (largest t) wins.
  if (tid < NT) { int p = s_bp[tid]; s_ov[p] = 2.0f; s_idx[p] = -1; }
  __syncthreads();
  if (tid < NT) { atomicMax(&s_idx[s_bp[tid]], tid); }
  __syncthreads();

  // ---- Phase 3: conf_t, 21-class argmax, per-wave histogram
  {
    const int w = tid >> 5;
    for (int p = tid; p < P_TOTAL; p += NTHREADS) {
      int ct = (s_ov[p] < 0.5f) ? 0 : ((int)s_lab[s_idx[p]] + 1);
      const float* row = conf + (size_t)p * NC;
      float bv = row[0]; int pred = 0;
      #pragma unroll
      for (int c = 1; c < NC; ++c) {
        float v = row[c];
        if (v > bv) { bv = v; pred = c; }   // first-wins
      }
      int s = (p < 2166) ? 0 : (p < 2766) ? 1 : (p < 2916) ? 2 :
              (p < 2970) ? 3 : (p < 2986) ? 4 : 5;
      bool eq  = (ct == pred);
      bool pos = (ct > 0);
      int base = w * 32 + s * 4;
      if (eq) atomicAdd(&s_cnt[base + 0], 1); else atomicAdd(&s_cnt[base + 1], 1);
      if (pos) {
        if (eq) atomicAdd(&s_cnt[base + 2], 1); else atomicAdd(&s_cnt[base + 3], 1);
      }
    }
  }
  __syncthreads();

  // ---- Phase 4: reduce 32 waves x 24 categories with the matrix pipe.
  // D = A(ones 16x32) x B(32x16) + 0, where B[k=wave][n=category] = count.
  // Per-wave counts <= 96 -> exact in f16; accumulate in f32 C.
  // B layout used here (contiguous k, not ISA table order): lane L holds N = L&15; half (L>>4) covers
  // K = 16*(L>>4) + {2j, 2j+1} in packed VGPR j. All waves run the WMMA
  // (EXEC all ones, no divergence); wave 0 lanes write the 24 outputs.
  {
    const int l     = tid & 31;
    const int lhalf = l >> 4;
    const int c0    = l & 15;
    v16h a, b1, b2;
    #pragma unroll
    for (int e = 0; e < 16; ++e) a[e] = (_Float16)1.0f;
    #pragma unroll
    for (int k = 0; k < 16; ++k) {
      int wv = lhalf * 16 + k;                    // K index == wave index
      b1[k] = (_Float16)(float)s_cnt[wv * 32 + c0];
      int c2 = 16 + c0;
      b2[k] = (c2 < 24) ? (_Float16)(float)s_cnt[wv * 32 + c2] : (_Float16)0.0f;
    }
    v8f acc1 = {}, acc2 = {};
    acc1 = __builtin_amdgcn_wmma_f32_16x16x32_f16(false, a, false, b1,
                                                  (short)0, acc1, false, false);
    acc2 = __builtin_amdgcn_wmma_f32_16x16x32_f16(false, a, false, b2,
                                                  (short)0, acc2, false, false);
    // D row 0: lanes 0..15 VGPR0 hold S[N=lane]
    if (tid < 16) out[tid]      = acc1[0];
    if (tid < 8)  out[16 + tid] = acc2[0];
  }
}

extern "C" void kernel_launch(void* const* d_in, const int* in_sizes, int n_in,
                              void* d_out, int out_size, void* d_ws, size_t ws_size,
                              hipStream_t stream) {
  (void)in_sizes; (void)n_in; (void)out_size; (void)d_ws; (void)ws_size;
  // input order: loc_data (unused), conf_data, priors, targets
  const float* conf    = (const float*)d_in[1];  // batch 0 = first 2990*21 floats
  const float* priors  = (const float*)d_in[2];
  const float* targets = (const float*)d_in[3];  // batch 0 = first 160 floats
  float* out = (float*)d_out;

  ssd_match_hist_kernel<<<dim3(1), dim3(NTHREADS), SMEM_BYTES, stream>>>(
      conf, priors, targets, out);
}
